// Attention_91302414778446
// MI455X (gfx1250) — compile-verified, matrix operand order checked by reading, not run on hardware
//
#include <hip/hip_runtime.h>

typedef __attribute__((ext_vector_type(16))) _Float16 v16h;
typedef __attribute__((ext_vector_type(8)))  _Float16 v8h;
typedef __attribute__((ext_vector_type(8)))  float    v8f;
typedef __attribute__((ext_vector_type(4)))  float    v4f;

#define IN_F   2048
#define OUT_F  2048
#define GROUP  32
#define M_TOK  16384   // 4 * 4096 tokens

// ---------------------------------------------------------------------------
// Async global->LDS copy (CDNA5). Tracked with ASYNCcnt; LDS written directly.
// LDS byte address = low 32 bits of the flat pointer (ISA 10.2 aperture rule).
// ---------------------------------------------------------------------------
__device__ __forceinline__ void async_load_b128(void* lds, const void* gaddr) {
    asm volatile("global_load_async_to_lds_b128 %0, %1, off"
                 :: "v"((unsigned int)(unsigned long long)lds), "v"(gaddr)
                 : "memory");
}

// ---------------------------------------------------------------------------
// Kernel 1: per-token asymmetric int8 fake-quant of activations -> f16
// ---------------------------------------------------------------------------
__global__ __launch_bounds__(256) void act_quant_f16(const float* __restrict__ x,
                                                     _Float16* __restrict__ xdq) {
    const int token = blockIdx.x;
    const int t     = threadIdx.x;
    const float* row = x + (size_t)token * IN_F;

    float v[8];
    {
        const v4f* rp = (const v4f*)(row + t * 8);
        v4f a = rp[0], b = rp[1];
        v[0] = a.x; v[1] = a.y; v[2] = a.z; v[3] = a.w;
        v[4] = b.x; v[5] = b.y; v[6] = b.z; v[7] = b.w;
    }

    float lmn = 0.0f, lmx = 0.0f;   // folded with 0 like the reference
#pragma unroll
    for (int i = 0; i < 8; ++i) { lmn = fminf(lmn, v[i]); lmx = fmaxf(lmx, v[i]); }

#pragma unroll
    for (int off = 16; off > 0; off >>= 1) {
        lmn = fminf(lmn, __shfl_xor(lmn, off, 32));
        lmx = fmaxf(lmx, __shfl_xor(lmx, off, 32));
    }

    __shared__ float smn[8], smx[8];
    const int wid = t >> 5, lane = t & 31;
    if (lane == 0) { smn[wid] = lmn; smx[wid] = lmx; }
    __syncthreads();

    float mn = smn[0], mx = smx[0];
#pragma unroll
    for (int i = 1; i < 8; ++i) { mn = fminf(mn, smn[i]); mx = fmaxf(mx, smx[i]); }

    const float eps   = 1.1920929e-7f;
    const float scale = fmaxf((mx - mn) * (1.0f / 255.0f), eps);
    const float inv_s = 1.0f / scale;
    const float zp = fminf(fmaxf(-128.0f - rintf(mn * inv_s), -128.0f), 127.0f);

    v8h outv;
#pragma unroll
    for (int i = 0; i < 8; ++i) {
        float q = fminf(fmaxf(rintf(v[i] * inv_s) + zp, -128.0f), 127.0f);
        outv[i] = (_Float16)((q - zp) * scale);
    }
    *(v8h*)(xdq + (size_t)token * IN_F + t * 8) = outv;
}

// ---------------------------------------------------------------------------
// Kernel 2: int4 (stored int32) group-dequant weights -> f16 row-major [O, I]
// ---------------------------------------------------------------------------
__global__ __launch_bounds__(256) void wt_dequant_f16(const int*   __restrict__ wq,
                                                      const float* __restrict__ wsc,
                                                      const float* __restrict__ wz,
                                                      _Float16*    __restrict__ wdq) {
    const size_t idx = ((size_t)blockIdx.x * 256 + threadIdx.x) * 4;
    const int o = (int)(idx / IN_F);
    const int i = (int)(idx % IN_F);
    const int g = i / GROUP;
    const float s = wsc[o * (IN_F / GROUP) + g];
    const float z = wz [o * (IN_F / GROUP) + g];
#pragma unroll
    for (int k = 0; k < 4; ++k)
        wdq[idx + k] = (_Float16)(((float)wq[idx + k] - z) * s);
}

// ---------------------------------------------------------------------------
// Kernel 3: f16 WMMA GEMM  C[M,N] = A[M,K] * B[N,K]^T  (f32 accumulate)
// block tile 128x128, K-step 32, double-buffered async global->LDS staging,
// 8 waves (4M x 2N), wave tile 32x64 -> 8 WMMA per K-step per wave.
// ---------------------------------------------------------------------------
#define BM 128
#define BN 128
#define BK 32
#define LDP 40                    // padded row stride (halfs): 80B, conflict-free
#define NSTEP (IN_F / BK)         // 64

__global__ __launch_bounds__(256) void gemm_f16_wmma(const _Float16* __restrict__ A,
                                                     const _Float16* __restrict__ B,
                                                     float* __restrict__ C) {
    __shared__ _Float16 sA[2][BM][LDP];
    __shared__ _Float16 sB[2][BN][LDP];

    const int t     = threadIdx.x;
    const int wid   = t >> 5;
    const int lane  = t & 31;
    const int r     = lane & 15;     // row within a 16x16 fragment
    const int hsel  = lane >> 4;     // K-half select (ISA 16-bit A/B layout)
    const int waveM = wid >> 1;      // 0..3 -> 32-row slice
    const int waveN = wid & 1;       // 0..1 -> 64-col slice

    const int bm = blockIdx.y * BM;
    const int bn = blockIdx.x * BN;

    v8f acc[2][4] = {};

    // staging: each thread owns 16 halfs of the A tile and 16 halfs of the B
    // tile per K-step -> 2+2 async b128 ops per tile
    const int srow = t >> 1, scol = (t & 1) * 16;
    const _Float16* gA = A + (size_t)(bm + srow) * IN_F + scol;
    const _Float16* gB = B + (size_t)(bn + srow) * IN_F + scol;

    auto issue = [&](int buf, int k0) {
        async_load_b128(&sA[buf][srow][scol],     gA + k0);
        async_load_b128(&sA[buf][srow][scol + 8], gA + k0 + 8);
        async_load_b128(&sB[buf][srow][scol],     gB + k0);
        async_load_b128(&sB[buf][srow][scol + 8], gB + k0 + 8);
    };

    issue(0, 0);
    issue(1, BK);

    for (int s = 0; s < NSTEP; ++s) {
        const int buf = s & 1;
        // oldest 4 async ops (this step's tile) must be complete; keep the
        // next tile's 4 ops in flight
        if (s + 1 < NSTEP) asm volatile("s_wait_asynccnt 0x4" ::: "memory");
        else               asm volatile("s_wait_asynccnt 0x0" ::: "memory");
        __syncthreads();

        v16h af[2], bf[4];
#pragma unroll
        for (int i = 0; i < 2; ++i) {
            const int ar = waveM * 32 + i * 16 + r;
            v8h lo = *(const v8h*)&sA[buf][ar][hsel * 8];
            v8h hi = *(const v8h*)&sA[buf][ar][16 + hsel * 8];
            af[i] = __builtin_shufflevector(lo, hi, 0,1,2,3,4,5,6,7,8,9,10,11,12,13,14,15);
        }
#pragma unroll
        for (int j = 0; j < 4; ++j) {
            const int br = waveN * 64 + j * 16 + r;
            v8h lo = *(const v8h*)&sB[buf][br][hsel * 8];
            v8h hi = *(const v8h*)&sB[buf][br][16 + hsel * 8];
            bf[j] = __builtin_shufflevector(lo, hi, 0,1,2,3,4,5,6,7,8,9,10,11,12,13,14,15);
        }
#pragma unroll
        for (int i = 0; i < 2; ++i)
#pragma unroll
            for (int j = 0; j < 4; ++j)
                acc[i][j] = __builtin_amdgcn_wmma_f32_16x16x32_f16(
                    false, af[i], false, bf[j],
                    (short)0, acc[i][j], false, false);

        __syncthreads();                       // everyone done reading `buf`
        if (s + 2 < NSTEP) issue(buf, (s + 2) * BK);  // refill behind compute
    }

    // epilogue: C/D layout -> M = hsel*8 + vgpr index, N = lane%16
#pragma unroll
    for (int i = 0; i < 2; ++i)
#pragma unroll
        for (int j = 0; j < 4; ++j) {
            const int m0 = bm + waveM * 32 + i * 16 + hsel * 8;
            const int n  = bn + waveN * 64 + j * 16 + r;
#pragma unroll
            for (int rr = 0; rr < 8; ++rr)
                C[(size_t)(m0 + rr) * OUT_F + n] = acc[i][j][rr];
        }
}

// ---------------------------------------------------------------------------
extern "C" void kernel_launch(void* const* d_in, const int* in_sizes, int n_in,
                              void* d_out, int out_size, void* d_ws, size_t ws_size,
                              hipStream_t stream) {
    const float* x   = (const float*)d_in[0];   // [4,4096,2048] f32
    const int*   wq  = (const int*)  d_in[1];   // [2048,2048] int32 (int4 values)
    const float* wsc = (const float*)d_in[2];   // [2048,64] f32
    const float* wz  = (const float*)d_in[3];   // [2048,64] f32
    float*       out = (float*)d_out;           // [4,4096,2048] f32

    _Float16* xdq = (_Float16*)d_ws;                       // 64 MB
    _Float16* wdq = xdq + (size_t)M_TOK * IN_F;            // +8 MB

    act_quant_f16<<<M_TOK, 256, 0, stream>>>(x, xdq);
    wt_dequant_f16<<<(OUT_F * IN_F) / 1024, 256, 0, stream>>>(wq, wsc, wz, wdq);

    dim3 grid(OUT_F / BN, M_TOK / BM);   // (16, 128)
    gemm_f16_wmma<<<grid, 256, 0, stream>>>(xdq, wdq, out);
}
